// MultiHeadAttn_FishPP_35699768165046
// MI455X (gfx1250) — matrix operand order repaired and checked by reading, not compiled, not run on hardware
//
#include <hip/hip_runtime.h>

#define N_TOK   197
#define BATCH   64
#define DMODEL  768
#define NHEAD   12
#define GHEAD   2
#define HRATIO  6
#define DHEAD   64
#define QKDIM   128
#define MROWS   (N_TOK * BATCH)   // 12608
#define QKV_COLS 1024             // 128 q + 128 k + 768 v
#define JPAD    224               // 197 padded up to multiple of 32

typedef __attribute__((ext_vector_type(16))) __bf16 v16bf;
typedef __attribute__((ext_vector_type(8)))  float  v8f;

union FragU { uint4 u[2]; v16bf v; };

__device__ __forceinline__ v8f wmma_bf16(v16bf a, v16bf b, v8f c) {
  // D = A(16x32 bf16) * B(32x16 bf16) + C(16x16 f32)
  return __builtin_amdgcn_wmma_f32_16x16x32_bf16(false, a, false, b, (short)0, c, false, false);
}

// A fragment 16x32 (MxK), LDS layout s[m][k], lda in elements.
// Lane l (0..15): M=l, K=0..7 & 16..23 ; lane 16..31: M=l-16, K=8..15 & 24..31.
__device__ __forceinline__ v16bf load_a_frag(const __bf16* s, int lda) {
  int lane = threadIdx.x & 31;
  int m  = lane & 15;
  int kb = (lane >> 4) << 3;          // 0 or 8
  FragU f;
  f.u[0] = *(const uint4*)(s + m * lda + kb);        // K = kb .. kb+7
  f.u[1] = *(const uint4*)(s + m * lda + kb + 16);   // K = kb+16 .. kb+23
  return f.v;
}

// B fragment 32x16 (KxN), LDS layout s[n][k] (element (k,n) at s[n*ldb+k]).
// Lane l (0..15): N=l, K=0..7 & 16..23 ; lane 16..31: N=l-16, K=8..15 & 24..31.
__device__ __forceinline__ v16bf load_b_frag(const __bf16* s, int ldb) {
  int lane = threadIdx.x & 31;
  int n  = lane & 15;
  int kb = (lane >> 4) << 3;          // 0 or 8
  FragU f;
  f.u[0] = *(const uint4*)(s + n * ldb + kb);
  f.u[1] = *(const uint4*)(s + n * ldb + kb + 16);
  return f.v;
}

// ---------------- small prep kernels ----------------

__global__ void k_f32_to_bf16(const float* __restrict__ in, __bf16* __restrict__ out, int n) {
  int i = blockIdx.x * blockDim.x + threadIdx.x;
  if (i < n) out[i] = (__bf16)in[i];
}

__global__ void k_pack_wcat(const float* __restrict__ Wq, const float* __restrict__ Wk,
                            const float* __restrict__ Wv, __bf16* __restrict__ wcat) {
  int i = blockIdx.x * blockDim.x + threadIdx.x;
  if (i >= QKV_COLS * DMODEL) return;
  int row = i / DMODEL, col = i % DMODEL;
  float v;
  if (row < 128)      v = Wq[row * DMODEL + col];
  else if (row < 256) v = Wk[(row - 128) * DMODEL + col];
  else                v = Wv[(row - 256) * DMODEL + col];
  wcat[i] = (__bf16)v;
}

__global__ void k_mask_mix(const float* __restrict__ masks, const float* __restrict__ mask_proj,
                           float* __restrict__ mw) {
  int i = blockIdx.x * blockDim.x + threadIdx.x;   // over N*N
  if (i >= N_TOK * N_TOK) return;
  float m0 = masks[i * 3 + 0], m1 = masks[i * 3 + 1], m2 = masks[i * 3 + 2];
#pragma unroll
  for (int h = 0; h < NHEAD; ++h)
    mw[(size_t)i * NHEAD + h] = m0 * mask_proj[0 * NHEAD + h]
                              + m1 * mask_proj[1 * NHEAD + h]
                              + m2 * mask_proj[2 * NHEAD + h];
}

// ---------------- main bf16 WMMA GEMM: C[M,NC] = A[M,K] * Bw[NC,K]^T ----------------
// Double-buffered LDS tiles: next tile's global loads are issued before the
// current tile's 8 WMMAs, hiding HBM latency; one barrier per K-step.
// Out-of-range M rows are clamped (never stored), so the hot loop has no
// exec-mask branching.

__global__ __launch_bounds__(256) void k_gemm_bf16(
    const __bf16* __restrict__ A, const __bf16* __restrict__ Bw,
    __bf16* __restrict__ Cb, float* __restrict__ Cf,
    int M, int K, int NC) {
  __shared__ __bf16 As[2][128][40];
  __shared__ __bf16 Bs[2][128][40];
  const int tid = threadIdx.x;
  const int wid = tid >> 5;
  const int m0 = blockIdx.x * 128;
  const int n0 = blockIdx.y * 128;
  const int wm = (wid >> 1) * 32;   // 0,32,64,96
  const int wn = (wid & 1) * 64;    // 0,64

  // Staging: tile = 128 rows x 32 cols bf16 = 128 rows x 4 uint4; 2 slots/thread.
  const int r0 = tid >> 2;              // 0..63
  const int c0 = (tid & 3) << 3;
  const int r1 = r0 + 64;               // 64..127
  const __bf16* A0 = A  + (size_t)min(m0 + r0, M - 1) * K + c0;
  const __bf16* A1 = A  + (size_t)min(m0 + r1, M - 1) * K + c0;
  const __bf16* B0 = Bw + (size_t)(n0 + r0) * K + c0;
  const __bf16* B1 = Bw + (size_t)(n0 + r1) * K + c0;

  const v8f vzero = {0.f,0.f,0.f,0.f,0.f,0.f,0.f,0.f};
  v8f acc[2][4];
#pragma unroll
  for (int i = 0; i < 2; ++i)
#pragma unroll
    for (int j = 0; j < 4; ++j) acc[i][j] = vzero;

  // prologue: stage k0 = 0
  uint4 sa0 = *(const uint4*)(A0);
  uint4 sa1 = *(const uint4*)(A1);
  uint4 sb0 = *(const uint4*)(B0);
  uint4 sb1 = *(const uint4*)(B1);
  *(uint4*)(&As[0][r0][c0]) = sa0;
  *(uint4*)(&As[0][r1][c0]) = sa1;
  *(uint4*)(&Bs[0][r0][c0]) = sb0;
  *(uint4*)(&Bs[0][r1][c0]) = sb1;
  __syncthreads();

  int buf = 0;
  for (int k0 = 0; k0 < K; k0 += 32) {
    const int kn = k0 + 32;
    const bool more = kn < K;
    if (more) {           // issue next tile's loads before computing
      sa0 = *(const uint4*)(A0 + kn);
      sa1 = *(const uint4*)(A1 + kn);
      sb0 = *(const uint4*)(B0 + kn);
      sb1 = *(const uint4*)(B1 + kn);
      if (kn + 32 < K) {  // L2 prefetch of the tile after next
        __builtin_prefetch(A0 + kn + 32, 0, 1);
        __builtin_prefetch(B0 + kn + 32, 0, 1);
      }
    }

    v16bf af[2], bfr[4];
#pragma unroll
    for (int i = 0; i < 2; ++i) af[i]  = load_a_frag(&As[buf][wm + i * 16][0], 40);
#pragma unroll
    for (int j = 0; j < 4; ++j) bfr[j] = load_b_frag(&Bs[buf][wn + j * 16][0], 40);
#pragma unroll
    for (int i = 0; i < 2; ++i)
#pragma unroll
      for (int j = 0; j < 4; ++j)
        acc[i][j] = wmma_bf16(af[i], bfr[j], acc[i][j]);

    if (more) {
      const int nb = buf ^ 1;
      *(uint4*)(&As[nb][r0][c0]) = sa0;
      *(uint4*)(&As[nb][r1][c0]) = sa1;
      *(uint4*)(&Bs[nb][r0][c0]) = sb0;
      *(uint4*)(&Bs[nb][r1][c0]) = sb1;
      __syncthreads();
      buf = nb;
    }
  }

  const int lane = tid & 31;
  const int cn = lane & 15;
  const int rb = (lane >> 4) * 8;
#pragma unroll
  for (int i = 0; i < 2; ++i)
#pragma unroll
    for (int j = 0; j < 4; ++j)
#pragma unroll
      for (int r = 0; r < 8; ++r) {
        int gm = m0 + wm + i * 16 + rb + r;
        int gn = n0 + wn + j * 16 + cn;
        if (gm < M) {
          float v = acc[i][j][r];
          if (Cb) Cb[(size_t)gm * NC + gn] = (__bf16)v;
          else    Cf[(size_t)gm * NC + gn] = v;
        }
      }
}

// ---------------- scores: S[b,g,i,j] = scale * sum_d Q[i,b,g,d] K[j,b,g,d] ----------------

__global__ __launch_bounds__(128) void k_score(const __bf16* __restrict__ qkv, float* __restrict__ S) {
  __shared__ __bf16 Qs[64][40];
  __shared__ __bf16 Ks[64][40];
  const int bg = blockIdx.z;            // b*2+g
  const int b = bg >> 1;
  const int g = bg & 1;
  const int i0 = blockIdx.x * 64;
  const int j0 = blockIdx.y * 64;
  const int tid = threadIdx.x;          // 128
  const int wid = tid >> 5;
  const int wm = (wid >> 1) * 32;
  const int wn = (wid & 1) * 32;
  const v8f vzero = {0.f,0.f,0.f,0.f,0.f,0.f,0.f,0.f};
  v8f acc[2][2];
#pragma unroll
  for (int i = 0; i < 2; ++i)
#pragma unroll
    for (int j = 0; j < 2; ++j) acc[i][j] = vzero;

  for (int k0 = 0; k0 < DHEAD; k0 += 32) {
#pragma unroll
    for (int i = 0; i < 2; ++i) {
      int lin = tid + i * 128;
      int r = lin >> 2;
      int c = (lin & 3) << 3;
      int gi = min(i0 + r, N_TOK - 1);   // clamped rows never stored
      int gj = min(j0 + r, N_TOK - 1);
      uint4 qv = *(const uint4*)(qkv + (size_t)(gi * BATCH + b) * QKV_COLS + g * DHEAD + k0 + c);
      uint4 kv = *(const uint4*)(qkv + (size_t)(gj * BATCH + b) * QKV_COLS + QKDIM + g * DHEAD + k0 + c);
      *(uint4*)(&Qs[r][c]) = qv;
      *(uint4*)(&Ks[r][c]) = kv;
    }
    __syncthreads();
    v16bf af[2], bfr[2];
#pragma unroll
    for (int i = 0; i < 2; ++i) af[i]  = load_a_frag(&Qs[wm + i * 16][0], 40);
#pragma unroll
    for (int j = 0; j < 2; ++j) bfr[j] = load_b_frag(&Ks[wn + j * 16][0], 40);
#pragma unroll
    for (int i = 0; i < 2; ++i)
#pragma unroll
      for (int j = 0; j < 2; ++j)
        acc[i][j] = wmma_bf16(af[i], bfr[j], acc[i][j]);
    __syncthreads();
  }

  const int lane = tid & 31;
  const int cn = lane & 15;
  const int rb = (lane >> 4) * 8;
#pragma unroll
  for (int i = 0; i < 2; ++i)
#pragma unroll
    for (int j = 0; j < 2; ++j)
#pragma unroll
      for (int r = 0; r < 8; ++r) {
        int gi = i0 + wm + i * 16 + rb + r;
        int gj = j0 + wn + j * 16 + cn;
        if (gi < N_TOK && gj < N_TOK)
          S[((size_t)bg * N_TOK + gi) * N_TOK + gj] = acc[i][j][r] * 0.125f;  // 1/sqrt(64)
      }
}

// ---------------- masked softmax -> bf16 probabilities (padded to JPAD) ----------------

__global__ __launch_bounds__(256) void k_softmax(const float* __restrict__ S, const float* __restrict__ mw,
                                                 __bf16* __restrict__ P) {
  const int bh = blockIdx.x;            // b*NHEAD+h
  const int b = bh / NHEAD;
  const int h = bh % NHEAD;
  const int g = h / HRATIO;
  __shared__ float red[256];
  const int t = threadIdx.x;
  const size_t srow = (size_t)(b * GHEAD + g) * N_TOK;

  for (int i = 0; i < N_TOK; ++i) {
    bool act = t < N_TOK;
    float x = 0.f;
    if (act) x = S[(srow + i) * N_TOK + t] * mw[((size_t)i * N_TOK + t) * NHEAD + h];
    red[t] = act ? x : -3.0e38f;
    __syncthreads();
    for (int s = 128; s > 0; s >>= 1) { if (t < s) red[t] = fmaxf(red[t], red[t + s]); __syncthreads(); }
    float mx = red[0];
    __syncthreads();
    float e = act ? __expf(x - mx) : 0.f;
    red[t] = e;
    __syncthreads();
    for (int s = 128; s > 0; s >>= 1) { if (t < s) red[t] += red[t + s]; __syncthreads(); }
    float inv = 1.f / red[0];
    __syncthreads();
    if (t < JPAD) P[((size_t)bh * N_TOK + i) * JPAD + t] = (__bf16)(e * inv);  // tail cols -> exact 0
  }
}

// ---------------- AV: vec[i,b,h*64+d] = sum_j P[b,h,i,j] * V[j,b,h,d] ----------------
// Double-buffered. V rows j>=197 are clamped; they multiply the exact-zero
// padded columns of P, contributing nothing.

__global__ __launch_bounds__(128) void k_av(const __bf16* __restrict__ P,
                                            const __bf16* __restrict__ qkv,
                                            __bf16* __restrict__ vec) {
  __shared__ __bf16 Ps[2][64][40];
  __shared__ __bf16 Vs[2][64][40];   // transposed: Vs[.][d][j]
  const int bh = blockIdx.y;
  const int b = bh / NHEAD;
  const int h = bh % NHEAD;
  const int i0 = blockIdx.x * 64;
  const int tid = threadIdx.x;    // 128
  const int wid = tid >> 5;
  const int wm = wid * 16;        // each wave: 16 rows x 64 cols
  const v8f vzero = {0.f,0.f,0.f,0.f,0.f,0.f,0.f,0.f};
  v8f acc[4];
#pragma unroll
  for (int j = 0; j < 4; ++j) acc[j] = vzero;

  // P staging: 64 rows x 4 uint4 -> 2 slots/thread
  const int pr0 = tid >> 2;                 // 0..31
  const int pc0 = (tid & 3) << 3;
  const int pr1 = pr0 + 32;                 // 32..63
  const __bf16* P0 = P + ((size_t)bh * N_TOK + min(i0 + pr0, N_TOK - 1)) * JPAD + pc0;
  const __bf16* P1 = P + ((size_t)bh * N_TOK + min(i0 + pr1, N_TOK - 1)) * JPAD + pc0;
  // V staging: 32 j-rows x 8 uint4 over d -> 2 slots/thread
  const int vj0 = tid >> 3;                 // 0..15
  const int vd0 = (tid & 7) << 3;
  const int vj1 = vj0 + 16;                 // 16..31
  const size_t vcol = 2 * QKDIM + (size_t)h * DHEAD + vd0;

  // prologue: j0 = 0
  uint4 sp0 = *(const uint4*)(P0);
  uint4 sp1 = *(const uint4*)(P1);
  uint4 sv0 = *(const uint4*)(qkv + (size_t)(min(vj0, N_TOK - 1) * BATCH + b) * QKV_COLS + vcol);
  uint4 sv1 = *(const uint4*)(qkv + (size_t)(min(vj1, N_TOK - 1) * BATCH + b) * QKV_COLS + vcol);
  *(uint4*)(&Ps[0][pr0][pc0]) = sp0;
  *(uint4*)(&Ps[0][pr1][pc0]) = sp1;
  {
    const __bf16* t0 = (const __bf16*)&sv0;
    const __bf16* t1 = (const __bf16*)&sv1;
#pragma unroll
    for (int e = 0; e < 8; ++e) { Vs[0][vd0 + e][vj0] = t0[e]; Vs[0][vd0 + e][vj1] = t1[e]; }
  }
  __syncthreads();

  int buf = 0;
  for (int j0 = 0; j0 < JPAD; j0 += 32) {
    const int jn = j0 + 32;
    const bool more = jn < JPAD;
    if (more) {
      sp0 = *(const uint4*)(P0 + jn);
      sp1 = *(const uint4*)(P1 + jn);
      sv0 = *(const uint4*)(qkv + (size_t)(min(jn + vj0, N_TOK - 1) * BATCH + b) * QKV_COLS + vcol);
      sv1 = *(const uint4*)(qkv + (size_t)(min(jn + vj1, N_TOK - 1) * BATCH + b) * QKV_COLS + vcol);
    }

    v16bf af = load_a_frag(&Ps[buf][wm][0], 40);
#pragma unroll
    for (int j = 0; j < 4; ++j) {
      v16bf bfr = load_b_frag(&Vs[buf][j * 16][0], 40);
      acc[j] = wmma_bf16(af, bfr, acc[j]);
    }

    if (more) {
      const int nb = buf ^ 1;
      *(uint4*)(&Ps[nb][pr0][pc0]) = sp0;
      *(uint4*)(&Ps[nb][pr1][pc0]) = sp1;
      const __bf16* t0 = (const __bf16*)&sv0;
      const __bf16* t1 = (const __bf16*)&sv1;
#pragma unroll
      for (int e = 0; e < 8; ++e) { Vs[nb][vd0 + e][vj0] = t0[e]; Vs[nb][vd0 + e][vj1] = t1[e]; }
      __syncthreads();
      buf = nb;
    }
  }

  const int lane = tid & 31;
  const int cn = lane & 15;
  const int rb = (lane >> 4) * 8;
#pragma unroll
  for (int j = 0; j < 4; ++j)
#pragma unroll
    for (int r = 0; r < 8; ++r) {
      int gi = i0 + wm + rb + r;
      int gd = j * 16 + cn;
      if (gi < N_TOK)
        vec[(size_t)(gi * BATCH + b) * DMODEL + h * DHEAD + gd] = (__bf16)acc[j][r];
    }
}

// ---------------- residual + LayerNorm ----------------

__global__ __launch_bounds__(256) void k_ln(const float* __restrict__ hin,
                                            const float* __restrict__ attn,
                                            const float* __restrict__ gamma,
                                            const float* __restrict__ beta,
                                            float* __restrict__ out) {
  const int row = blockIdx.x;
  const int t = threadIdx.x;
  __shared__ float red[256];
  __shared__ float stat[2];
  const float* hr = hin + (size_t)row * DMODEL;
  const float* ar = attn + (size_t)row * DMODEL;
  float xv[3];
  float s = 0.f;
#pragma unroll
  for (int q = 0; q < 3; ++q) {
    int c = t + q * 256;
    float x = hr[c] + ar[c];
    xv[q] = x;
    s += x;
  }
  red[t] = s; __syncthreads();
  for (int st = 128; st > 0; st >>= 1) { if (t < st) red[t] += red[t + st]; __syncthreads(); }
  if (t == 0) stat[0] = red[0] * (1.f / DMODEL);
  __syncthreads();
  float mu = stat[0];
  float s2 = 0.f;
#pragma unroll
  for (int q = 0; q < 3; ++q) { float d = xv[q] - mu; s2 += d * d; }
  red[t] = s2; __syncthreads();
  for (int st = 128; st > 0; st >>= 1) { if (t < st) red[t] += red[t + st]; __syncthreads(); }
  if (t == 0) stat[1] = red[0] * (1.f / DMODEL);
  __syncthreads();
  float rinv = rsqrtf(stat[1] + 1e-5f);
#pragma unroll
  for (int q = 0; q < 3; ++q) {
    int c = t + q * 256;
    out[(size_t)row * DMODEL + c] = gamma[c] * (xv[q] - mu) * rinv + beta[c];
  }
}

// ---------------- launch ----------------

extern "C" void kernel_launch(void* const* d_in, const int* in_sizes, int n_in,
                              void* d_out, int out_size, void* d_ws, size_t ws_size,
                              hipStream_t stream) {
  const float* h         = (const float*)d_in[0];
  const float* masks     = (const float*)d_in[1];
  const float* Wq        = (const float*)d_in[2];
  const float* Wk        = (const float*)d_in[3];
  const float* Wv        = (const float*)d_in[4];
  const float* Wo        = (const float*)d_in[5];
  const float* mask_proj = (const float*)d_in[6];
  const float* gamma     = (const float*)d_in[7];
  const float* beta      = (const float*)d_in[8];
  float* out = (float*)d_out;
  (void)in_sizes; (void)n_in; (void)out_size; (void)ws_size;

  char* ws = (char*)d_ws;
  size_t off = 0;
  auto take = [&](size_t bytes) { char* p = ws + off; off += (bytes + 255) & ~(size_t)255; return p; };

  __bf16* hb   = (__bf16*)take((size_t)MROWS * DMODEL * 2);
  __bf16* wcat = (__bf16*)take((size_t)QKV_COLS * DMODEL * 2);
  __bf16* wob  = (__bf16*)take((size_t)DMODEL * DMODEL * 2);
  __bf16* qkv  = (__bf16*)take((size_t)MROWS * QKV_COLS * 2);
  float*  Sbuf = (float*) take((size_t)BATCH * GHEAD * N_TOK * N_TOK * 4);
  float*  mwb  = (float*) take((size_t)N_TOK * N_TOK * NHEAD * 4);
  __bf16* Pbuf = (__bf16*)take((size_t)BATCH * NHEAD * N_TOK * JPAD * 2);
  __bf16* vec  = (__bf16*)take((size_t)MROWS * DMODEL * 2);
  float*  attn = (float*) take((size_t)MROWS * DMODEL * 4);

  int nh = MROWS * DMODEL;
  k_f32_to_bf16<<<(nh + 255) / 256, 256, 0, stream>>>(h, hb, nh);
  int nw = DMODEL * DMODEL;
  k_f32_to_bf16<<<(nw + 255) / 256, 256, 0, stream>>>(Wo, wob, nw);
  int nc = QKV_COLS * DMODEL;
  k_pack_wcat<<<(nc + 255) / 256, 256, 0, stream>>>(Wq, Wk, Wv, wcat);
  int nm = N_TOK * N_TOK;
  k_mask_mix<<<(nm + 255) / 256, 256, 0, stream>>>(masks, mask_proj, mwb);

  // QKV projection: [12608,768] x [1024,768]^T
  k_gemm_bf16<<<dim3((MROWS + 127) / 128, QKV_COLS / 128), 256, 0, stream>>>(
      hb, wcat, qkv, nullptr, MROWS, DMODEL, QKV_COLS);

  // scores per (b,g)
  k_score<<<dim3(4, 4, BATCH * GHEAD), 128, 0, stream>>>(qkv, Sbuf);

  // masked softmax per (b,h)
  k_softmax<<<BATCH * NHEAD, 256, 0, stream>>>(Sbuf, mwb, Pbuf);

  // AV per (b,h)
  k_av<<<dim3(4, BATCH * NHEAD), 128, 0, stream>>>(Pbuf, qkv, vec);

  // output projection: [12608,768] x [768,768]^T
  k_gemm_bf16<<<dim3((MROWS + 127) / 128, DMODEL / 128), 256, 0, stream>>>(
      vec, wob, nullptr, attn, MROWS, DMODEL, DMODEL);

  // residual + LayerNorm
  k_ln<<<MROWS, 256, 0, stream>>>(h, attn, gamma, beta, out);
}
